// Attention_28802050687686
// MI455X (gfx1250) — matrix operand order repaired and checked by reading, not compiled, not run on hardware
//
#include <hip/hip_runtime.h>
#include <hip/hip_bf16.h>

// ---------------------------------------------------------------------------
// CDNA5 (gfx1250) attention pipeline, wave32 + v_wmma_f32_16x16x32_bf16
// + global_load_async_to_lds_b128 (ASYNCcnt) staging.
// Shapes: B=1, T=4096, D=2048, N=8 q-heads, K=4 kv-heads, H=256, window=1024.
// ---------------------------------------------------------------------------

typedef __bf16 bf16_t;
typedef __attribute__((ext_vector_type(16))) __bf16 v16bf;
typedef __attribute__((ext_vector_type(8)))  float  v8f;

#define SEQ_T   4096
#define EMB_D   2048
#define NQ      8
#define NKV     4
#define HD      256
#define QKV_C   4096     // 8*256 + 4*256 + 4*256
#define WINDOW  1024
#define SOFTCAP 50.0f
#define QSCALE  0.0625f  // 256^-0.5

union FragBF {
    v16bf v;
    uint4 u[2];
};

__device__ __forceinline__ v8f vzero8() {
    v8f z = {0.f, 0.f, 0.f, 0.f, 0.f, 0.f, 0.f, 0.f};
    return z;
}

__device__ __forceinline__ v8f wmma_bf16(v16bf a, v16bf b, v8f c) {
    // D = A(16x32 bf16) * B(32x16 bf16) + C(16x16 f32)
    return __builtin_amdgcn_wmma_f32_16x16x32_bf16(
        false, a, false, b, (short)0, c, false, false);
}

// Async DMA: 16B global -> LDS per lane; tracked by ASYNCcnt.
// LDS operand = wave-relative byte address = low 32 bits of the generic pointer
// (AS3->generic addrspacecast is {aperture_hi, offset32} on amdgcn).
__device__ __forceinline__ void async_ld_b128(void* lds_dst, const void* gsrc) {
    asm volatile("global_load_async_to_lds_b128 %0, %1, off"
                 :: "v"((unsigned)(unsigned long long)lds_dst),
                    "v"((unsigned long long)gsrc)
                 : "memory");
}
__device__ __forceinline__ void wait_async0() {
    asm volatile("s_wait_asynccnt 0x0" ::: "memory");
}

// A fragment: 16x32, storage row-major [m][k], lane = M, two b128 loads.
// lane l: m = l&15, half = l>>4; elems 0..7 -> k = 8*half+e; 8..15 -> k = 16+8*half+(e-8)
__device__ __forceinline__ v16bf load_frag_a(const bf16_t* base, int ld, int lane) {
    int m = lane & 15, hl = lane >> 4;
    const bf16_t* p = base + m * ld + hl * 8;
    FragBF f;
    f.u[0] = *(const uint4*)(p);
    f.u[1] = *(const uint4*)(p + 16);
    return f.v;
}

// B fragment: 32x16 (KxN), storage [n][k] (k contiguous, stride ld), lane = N.
// lane l: n = l&15, half = l>>4; elems 0..7 -> k = 8*half+e; 8..15 -> k = 16+8*half+(e-8)
__device__ __forceinline__ v16bf load_frag_b(const bf16_t* base, int ld, int lane) {
    int n = lane & 15, hl = lane >> 4;
    const bf16_t* p = base + n * ld + hl * 8;
    FragBF f;
    f.u[0] = *(const uint4*)(p);
    f.u[1] = *(const uint4*)(p + 16);
    return f.v;
}

// ---------------------------------------------------------------------------
// Kernel 1: fused QKV projection GEMM: qkv[T][4096] = x[T][2048] * W[2048][4096]
//   cols 0..2047  : Q head n=c/256, h=c%256   -> q_w[n][d][h]
//   cols 2048..   : K/V kv_w[c2][kh][d][h]
// 64x128 block tile, 8 waves, each wave 32x32; K-step 64 => 8 WMMA per barrier.
// ---------------------------------------------------------------------------
__global__ __launch_bounds__(256) void qkv_gemm(const float* __restrict__ x,
                                                const float* __restrict__ q_w,
                                                const float* __restrict__ kv_w,
                                                float* __restrict__ qkv) {
    __shared__ bf16_t At[64 * 64];    // 8 KB
    __shared__ bf16_t Bt[128 * 64];   // 16 KB
    const int t0 = blockIdx.x * 64;
    const int c0 = blockIdx.y * 128;
    const int tid = threadIdx.x;
    const int lane = tid & 31, wave = tid >> 5;
    const int mi = (wave >> 2) * 32;   // 0 / 32
    const int ni = (wave & 3) * 32;    // 0..96

    v8f acc[2][2];
    acc[0][0] = vzero8(); acc[0][1] = vzero8();
    acc[1][0] = vzero8(); acc[1][1] = vzero8();

    for (int k0 = 0; k0 < EMB_D; k0 += 64) {
        __syncthreads();
        if (k0 + 128 < EMB_D)
            __builtin_prefetch(x + (size_t)t0 * EMB_D + k0 + 128, 0, 1); // global_prefetch_b8
        // stage A tile (x -> bf16), [m][k] k-contiguous
        for (int i = tid; i < 64 * 64; i += 256) {
            int m = i >> 6, kk = i & 63;
            At[i] = (bf16_t)x[(size_t)(t0 + m) * EMB_D + k0 + kk];
        }
        // stage B tile [c][k] (k contiguous); c fast over threads -> coalesced h
        for (int i = tid; i < 128 * 64; i += 256) {
            int c = i & 127, kk = i >> 7;
            int col = c0 + c, kd = k0 + kk;
            float wv;
            if (col < 2048) {
                int n = col >> 8, h = col & 255;
                wv = q_w[((size_t)n * EMB_D + kd) * HD + h];
            } else {
                int cc = col - 2048;
                int c2 = cc >> 10, kh = (cc >> 8) & 3, h = cc & 255;
                wv = kv_w[(((size_t)(c2 * NKV + kh)) * EMB_D + kd) * HD + h];
            }
            Bt[c * 64 + kk] = (bf16_t)wv;
        }
        __syncthreads();
        for (int kc = 0; kc < 64; kc += 32) {
            v16bf a0 = load_frag_a(At + (mi + 0) * 64 + kc, 64, lane);
            v16bf a1 = load_frag_a(At + (mi + 16) * 64 + kc, 64, lane);
            v16bf b0 = load_frag_b(Bt + (ni + 0) * 64 + kc, 64, lane);
            v16bf b1 = load_frag_b(Bt + (ni + 16) * 64 + kc, 64, lane);
            acc[0][0] = wmma_bf16(a0, b0, acc[0][0]);
            acc[0][1] = wmma_bf16(a0, b1, acc[0][1]);
            acc[1][0] = wmma_bf16(a1, b0, acc[1][0]);
            acc[1][1] = wmma_bf16(a1, b1, acc[1][1]);
        }
    }
    const int hl = lane >> 4, ln = lane & 15;
    for (int ii = 0; ii < 2; ++ii)
        for (int jj = 0; jj < 2; ++jj)
            for (int r = 0; r < 8; ++r) {
                int row = t0 + mi + 16 * ii + r + 8 * hl;
                int col = c0 + ni + 16 * jj + ln;
                qkv[(size_t)row * QKV_C + col] = acc[ii][jj][r];
            }
}

// ---------------------------------------------------------------------------
// Kernel 2: RMSNorm (+scale) + RoPE (+ H^-0.5 on Q), f32 -> bf16.
// grid (T, 16): y = 0..7 Q-head, 8..11 K-head, 12..15 V-head. 256 threads.
// ---------------------------------------------------------------------------
__global__ __launch_bounds__(256) void norm_rope(const float* __restrict__ qkv,
                                                 const float* __restrict__ q_scale,
                                                 const float* __restrict__ k_scale,
                                                 const float* __restrict__ v_scale,
                                                 const int* __restrict__ segment_pos,
                                                 bf16_t* __restrict__ Qb,
                                                 bf16_t* __restrict__ Kb,
                                                 bf16_t* __restrict__ Vb) {
    __shared__ float buf[HD];
    __shared__ float red[8];
    const int t = blockIdx.x;
    const int which = blockIdx.y;
    const int h = threadIdx.x;

    int col; float sc;
    if (which < 8)       { col = which * HD + h;                sc = q_scale[h]; }
    else if (which < 12) { col = 2048 + (which - 8) * HD + h;   sc = k_scale[h]; }
    else                 { col = 3072 + (which - 12) * HD + h;  sc = v_scale[h]; }

    float val = qkv[(size_t)t * QKV_C + col];
    float sq = val * val;
    for (int off = 16; off > 0; off >>= 1) sq += __shfl_xor(sq, off);
    if ((threadIdx.x & 31) == 0) red[threadIdx.x >> 5] = sq;
    __syncthreads();
    float tot = red[0] + red[1] + red[2] + red[3] + red[4] + red[5] + red[6] + red[7];
    float rinv = rsqrtf(tot * (1.0f / HD) + 1e-6f);
    float normed = val * rinv * (1.0f + sc);
    buf[h] = normed;
    __syncthreads();

    float out = normed;
    if (which < 12) {   // RoPE on Q and K only
        int pos = segment_pos[t];
        int i = h & 127;
        float ts = __powf(10000.0f, (2.0f * (float)i) * (1.0f / HD));
        float ang = (float)pos / ts;
        float s, c;
        __sincosf(ang, &s, &c);
        float x1 = buf[i], x2 = buf[i + 128];
        out = (h < 128) ? (x1 * c - x2 * s) : (x2 * c + x1 * s);
    }
    if (which < 8) out *= QSCALE;   // fold H^-0.5 into Q

    if (which < 8)       Qb[((size_t)t * NQ  + which)        * HD + h] = (bf16_t)out;
    else if (which < 12) Kb[((size_t)t * NKV + (which - 8))  * HD + h] = (bf16_t)out;
    else                 Vb[((size_t)t * NKV + (which - 12)) * HD + h] = (bf16_t)out;
}

// ---------------------------------------------------------------------------
// Kernel 3: sliding-window flash attention (GQA, tanh soft-cap).
// Block = 4 waves, 64 query rows, one q-head. Key tiles of 32.
// Q and K tiles staged with async DMA (global_load_async_to_lds_b128).
// ---------------------------------------------------------------------------
union FlashSmem {
    bf16_t qs[64 * HD];                                  // 32 KB Q staging
    struct { bf16_t kt[32 * HD]; bf16_t vt[HD * 32]; } kv; // 16+16 KB (reused)
};

__global__ __launch_bounds__(128) void flash_attn(const bf16_t* __restrict__ Qb,
                                                  const bf16_t* __restrict__ Kb,
                                                  const bf16_t* __restrict__ Vb,
                                                  bf16_t* __restrict__ enc) {
    __shared__ FlashSmem sm;
    __shared__ bf16_t Pl[4][16 * 32];   // per-wave P scratch

    const int t0 = blockIdx.x * 64;
    const int head = blockIdx.y;
    const int kvh = head >> 1;          // G = 2
    const int tid = threadIdx.x;
    const int lane = tid & 31, wave = tid >> 5;
    const int hl = lane >> 4, ln = lane & 15;

    // stage Q tile [64][256] bf16 via async DMA (64 rows x 512B = 2048 x 16B)
    for (int i = tid; i < 64 * 32; i += 128) {
        int m = i >> 5, w = i & 31;
        async_ld_b128((char*)sm.qs + i * 16,
                      (const char*)(Qb + (((size_t)(t0 + m)) * NQ + head) * HD) + w * 16);
    }
    wait_async0();
    __syncthreads();
    v16bf qf[8];
    for (int kc = 0; kc < 8; ++kc)
        qf[kc] = load_frag_a(sm.qs + (wave * 16) * HD + kc * 32, HD, lane);

    v8f o[16];
    for (int ct = 0; ct < 16; ++ct) o[ct] = vzero8();
    float m_r[8], l_r[8];
    for (int r = 0; r < 8; ++r) { m_r[r] = -60.0f; l_r[r] = 0.0f; }

    int jstart = t0 - (WINDOW - 1); if (jstart < 0) jstart = 0;
    jstart &= ~31;
    const int jend = t0 + 64;

    for (int j0 = jstart; j0 < jend; j0 += 32) {
        __syncthreads();   // protect sm (incl. Q region on first iter) + Pl reuse
        // stage K tile [key][kd] via async DMA (32 rows x 512B = 1024 x 16B)
        for (int i = tid; i < 32 * 32; i += 128) {
            int key = i >> 5, w = i & 31;
            async_ld_b128((char*)sm.kv.kt + i * 16,
                          (const char*)(Kb + (((size_t)(j0 + key)) * NKV + kvh) * HD)
                              + w * 16);
        }
        // stage V tile transposed: vt[h][key] (element-wise, cannot DMA-transpose)
        for (int i = tid; i < 32 * HD; i += 128) {
            int key = i >> 8, h = i & 255;
            sm.kv.vt[h * 32 + key] =
                Vb[(((size_t)(j0 + key)) * NKV + kvh) * HD + h];
        }
        wait_async0();
        __syncthreads();

        // S = Q * K^T  (two 16-key column sub-tiles)
        v8f s[2];
        for (int sub = 0; sub < 2; ++sub) {
            v8f sf = vzero8();
            for (int kc = 0; kc < 8; ++kc) {
                v16bf bf = load_frag_b(sm.kv.kt + (sub * 16) * HD + kc * 32, HD, lane);
                sf = wmma_bf16(qf[kc], bf, sf);
            }
            // soft-cap + causal/sliding-window mask (C layout: M=r+8*hl, N=ln)
            for (int r = 0; r < 8; ++r) {
                float v = tanhf(sf[r] * (1.0f / SOFTCAP)) * SOFTCAP;
                int row = t0 + wave * 16 + r + 8 * hl;
                int colj = j0 + sub * 16 + ln;
                bool valid = (colj <= row) && ((row - colj) < WINDOW);
                sf[r] = valid ? v : -3.0e38f;
            }
            s[sub] = sf;
        }

        // online softmax per row (16-lane row groups -> shfl_xor 1..8)
        float corr[8];
        for (int r = 0; r < 8; ++r) {
            float mx = fmaxf(s[0][r], s[1][r]);
            for (int off = 8; off > 0; off >>= 1) mx = fmaxf(mx, __shfl_xor(mx, off));
            float mnew = fmaxf(m_r[r], mx);
            corr[r] = __expf(m_r[r] - mnew);
            float p0 = __expf(s[0][r] - mnew);
            float p1 = __expf(s[1][r] - mnew);
            s[0][r] = p0; s[1][r] = p1;
            float rs = p0 + p1;
            for (int off = 8; off > 0; off >>= 1) rs += __shfl_xor(rs, off);
            l_r[r] = l_r[r] * corr[r] + rs;
            m_r[r] = mnew;
        }
        for (int ct = 0; ct < 16; ++ct)
            for (int r = 0; r < 8; ++r) o[ct][r] *= corr[r];

        // P -> per-wave LDS (same-wave LDS ops are in-order), reload as A frag
        for (int sub = 0; sub < 2; ++sub)
            for (int r = 0; r < 8; ++r)
                Pl[wave][(r + 8 * hl) * 32 + sub * 16 + ln] = (bf16_t)s[sub][r];
        v16bf pa = load_frag_a(Pl[wave], 32, lane);

        // O += P * V
        for (int ct = 0; ct < 16; ++ct) {
            v16bf bv = load_frag_b(sm.kv.vt + (ct * 16) * 32, 32, lane);
            o[ct] = wmma_bf16(pa, bv, o[ct]);
        }
    }

    float invl[8];
    for (int r = 0; r < 8; ++r) invl[r] = (l_r[r] > 0.0f) ? (1.0f / l_r[r]) : 0.0f;
    for (int ct = 0; ct < 16; ++ct)
        for (int r = 0; r < 8; ++r) {
            int row = t0 + wave * 16 + r + 8 * hl;
            int col = ct * 16 + ln;
            enc[(size_t)row * EMB_D + head * HD + col] = (bf16_t)(o[ct][r] * invl[r]);
        }
}

// ---------------------------------------------------------------------------
// Kernel 4: output projection: out[T][2048] = enc[T][2048](bf16) * o_w[2048][2048]
// (o_w[n][h][d] is exactly row-major [n*256+h][d]). Same tiling as kernel 1.
// ---------------------------------------------------------------------------
__global__ __launch_bounds__(256) void out_gemm(const bf16_t* __restrict__ enc,
                                                const float* __restrict__ o_w,
                                                float* __restrict__ out) {
    __shared__ bf16_t At[64 * 64];
    __shared__ bf16_t Bt[128 * 64];
    const int t0 = blockIdx.x * 64;
    const int c0 = blockIdx.y * 128;
    const int tid = threadIdx.x;
    const int lane = tid & 31, wave = tid >> 5;
    const int mi = (wave >> 2) * 32;
    const int ni = (wave & 3) * 32;

    v8f acc[2][2];
    acc[0][0] = vzero8(); acc[0][1] = vzero8();
    acc[1][0] = vzero8(); acc[1][1] = vzero8();

    for (int k0 = 0; k0 < EMB_D; k0 += 64) {
        __syncthreads();
        if (k0 + 128 < EMB_D)
            __builtin_prefetch(o_w + ((size_t)(k0 + 128)) * EMB_D + c0, 0, 1);
        for (int i = tid; i < 64 * 64; i += 256) {
            int m = i >> 6, kk = i & 63;
            At[i] = enc[(size_t)(t0 + m) * EMB_D + k0 + kk];
        }
        for (int i = tid; i < 128 * 64; i += 256) {
            int c = i & 127, kk = i >> 7;
            Bt[c * 64 + kk] = (bf16_t)o_w[((size_t)(k0 + kk)) * EMB_D + c0 + c];
        }
        __syncthreads();
        for (int kc = 0; kc < 64; kc += 32) {
            v16bf a0 = load_frag_a(At + (mi + 0) * 64 + kc, 64, lane);
            v16bf a1 = load_frag_a(At + (mi + 16) * 64 + kc, 64, lane);
            v16bf b0 = load_frag_b(Bt + (ni + 0) * 64 + kc, 64, lane);
            v16bf b1 = load_frag_b(Bt + (ni + 16) * 64 + kc, 64, lane);
            acc[0][0] = wmma_bf16(a0, b0, acc[0][0]);
            acc[0][1] = wmma_bf16(a0, b1, acc[0][1]);
            acc[1][0] = wmma_bf16(a1, b0, acc[1][0]);
            acc[1][1] = wmma_bf16(a1, b1, acc[1][1]);
        }
    }
    const int hl = lane >> 4, ln = lane & 15;
    for (int ii = 0; ii < 2; ++ii)
        for (int jj = 0; jj < 2; ++jj)
            for (int r = 0; r < 8; ++r) {
                int row = t0 + mi + 16 * ii + r + 8 * hl;
                int col = c0 + ni + 16 * jj + ln;
                out[(size_t)row * EMB_D + col] = acc[ii][jj][r];
            }
}

// ---------------------------------------------------------------------------
// Workspace layout (bytes):
//   [0,        67108864) qkv f32  [T][4096]      (reused as enc bf16 later)
//   [67108864, 83886080) Qb bf16  [T][8][256]
//   [83886080, 92274688) Kb bf16  [T][4][256]
//   [92274688,100663296) Vb bf16  [T][4][256]
// Total: 100,663,296 bytes.
// ---------------------------------------------------------------------------
extern "C" void kernel_launch(void* const* d_in, const int* in_sizes, int n_in,
                              void* d_out, int out_size, void* d_ws, size_t ws_size,
                              hipStream_t stream) {
    const float* x        = (const float*)d_in[0];
    const float* q_w      = (const float*)d_in[1];
    const float* kv_w     = (const float*)d_in[2];
    const float* o_w      = (const float*)d_in[3];
    const float* q_scale  = (const float*)d_in[4];
    const float* k_scale  = (const float*)d_in[5];
    const float* v_scale  = (const float*)d_in[6];
    const int*   seg_pos  = (const int*)d_in[7];
    // d_in[8] attn_mask: causal; reproduced analytically in flash_attn.
    (void)in_sizes; (void)n_in; (void)out_size; (void)ws_size;

    char* ws = (char*)d_ws;
    float*  qkv = (float*)ws;
    bf16_t* Qb  = (bf16_t*)(ws + 67108864);
    bf16_t* Kb  = (bf16_t*)(ws + 83886080);
    bf16_t* Vb  = (bf16_t*)(ws + 92274688);
    bf16_t* enc = (bf16_t*)ws;              // aliases qkv (dead after norm_rope)
    float*  out = (float*)d_out;

    qkv_gemm <<<dim3(SEQ_T / 64, QKV_C / 128), 256, 0, stream>>>(x, q_w, kv_w, qkv);
    norm_rope<<<dim3(SEQ_T, 16),               256, 0, stream>>>(qkv, q_scale, k_scale,
                                                                 v_scale, seg_pos,
                                                                 Qb, Kb, Vb);
    flash_attn<<<dim3(SEQ_T / 64, NQ),         128, 0, stream>>>(Qb, Kb, Vb, enc);
    out_gemm <<<dim3(SEQ_T / 64, EMB_D / 128), 256, 0, stream>>>(enc, o_w, out);
}
